// MPNN_44573170598880
// MI455X (gfx1250) — matrix operand order repaired and checked by reading, not compiled, not run on hardware
//
#include <hip/hip_runtime.h>

// ---------------------------------------------------------------------------
// MPNN / MetaLayer GNN for MI455X (gfx1250), wave32 + WMMA bf16.
//
// Pipeline per layer:
//   build A_e (bf16 gather-concat) -> WMMA GEMM -> ea
//   build A_m -> WMMA GEMM -> msg -> atomic scatter-sum -> agg
//   build A_n -> WMMA GEMM -> x
//   nagg (sorted-batch sums) + eagg (atomics) -> build A_g -> WMMA GEMM -> u
//   layers 0..2: training-mode BatchNorm on x / ea / u
// ---------------------------------------------------------------------------

#define CDIV(a, b) (((a) + (b) - 1) / (b))

typedef __attribute__((ext_vector_type(16))) __bf16 v16bf;
typedef __attribute__((ext_vector_type(8)))  float  v8f;

union Frag16 { v16bf v; unsigned u[8]; };

__device__ __forceinline__ unsigned short f2bf(float f) {
  unsigned u = __float_as_uint(f);
  u += 0x7FFFu + ((u >> 16) & 1u);   // round-to-nearest-even
  return (unsigned short)(u >> 16);
}

// ---------------------------------------------------------------------------
// Generic GEMM:  C(MxN, f32) = A(MxK, bf16 row-major) * Bt(NxK, bf16 n-major)^T
//                 + bias, optional ReLU.
// Block tile 128x128, 8 waves (4 M x 2 N), wave tile 32x64 = 2x4 WMMA 16x16x32.
// Per K-step per wave: 8 v_wmma vs 12 ds_load_b128.
// ---------------------------------------------------------------------------
__global__ __launch_bounds__(256)
void gemm_bf16_wmma(const unsigned short* __restrict__ A,
                    const unsigned short* __restrict__ Bt,
                    const float* __restrict__ bias,
                    float* __restrict__ C,
                    int M, int Nc, int K, int relu)
{
  __shared__ unsigned short sA[128][32];   // 8 KB
  __shared__ unsigned short sB[128][32];   // 8 KB (n-major)

  const int m0 = blockIdx.y * 128;
  const int n0 = blockIdx.x * 128;
  const int tid  = threadIdx.x;
  const int lane = tid & 31;
  const int wave = tid >> 5;
  const int wm = wave & 3;       // 0..3 (M direction, 32 rows each)
  const int wn = wave >> 2;      // 0..1 (N direction, 64 cols each)

  v8f acc[2][4] = {};

  // ISA 7.12.2 fragment addressing (wave32):
  // A 16x32: lane<16 -> M=lane, K in {0..7, 16..23}; lane>=16 -> K+8
  const int mA   = lane & 15;
  const int aoff = (lane & 16) ? 8 : 0;
  // B 32x16 staged n-major: lane<16 -> N=lane, K in {0..7, 16..23}; lane>=16 -> K+8
  const int nB   = lane & 15;
  const int boff = (lane & 16) ? 8 : 0;

  // staging assignment: 2 x uint4 (8 bf16) per thread for each of A and B
  const int str = tid >> 1;            // rows 0..127 (2 threads/row)
  const int stc = (tid & 1) << 4;      // col 0 or 16

  for (int k0 = 0; k0 < K; k0 += 32) {
    // stage A tile: 128x32 bf16
    {
      uint4 v0 = {0u, 0u, 0u, 0u}, v1 = v0;
      int gr = m0 + str;
      if (gr < M) {
        const unsigned short* p = &A[(size_t)gr * K + k0 + stc];
        v0 = *(const uint4*)p;
        v1 = *(const uint4*)(p + 8);
        if (k0 + 64 < K) __builtin_prefetch(p + 64, 0, 1);   // global_prefetch_b8
      }
      *(uint4*)&sA[str][stc]     = v0;
      *(uint4*)&sA[str][stc + 8] = v1;
    }
    // stage B tile: 128x32 bf16 (rows are output columns)
    {
      uint4 v0 = {0u, 0u, 0u, 0u}, v1 = v0;
      int gn = n0 + str;
      if (gn < Nc) {
        const unsigned short* p = &Bt[(size_t)gn * K + k0 + stc];
        v0 = *(const uint4*)p;
        v1 = *(const uint4*)(p + 8);
      }
      *(uint4*)&sB[str][stc]     = v0;
      *(uint4*)&sB[str][stc + 8] = v1;
    }
    __syncthreads();

    Frag16 afr[2], bfr[4];
#pragma unroll
    for (int r = 0; r < 2; ++r) {
      int row = wm * 32 + r * 16 + mA;
#pragma unroll
      for (int p = 0; p < 8; ++p) {
        int k = ((p < 4) ? (p * 2) : (16 + (p - 4) * 2)) + aoff;
        afr[r].u[p] = *(const unsigned*)&sA[row][k];
      }
    }
#pragma unroll
    for (int c = 0; c < 4; ++c) {
      int col = wn * 64 + c * 16 + nB;
#pragma unroll
      for (int p = 0; p < 8; ++p)
        bfr[c].u[p] = *(const unsigned*)&sB[col][((p < 4) ? (p * 2) : (16 + (p - 4) * 2)) + boff];
    }

#pragma unroll
    for (int r = 0; r < 2; ++r)
#pragma unroll
      for (int c = 0; c < 4; ++c)
        acc[r][c] = __builtin_amdgcn_wmma_f32_16x16x32_bf16(
            false, afr[r].v, false, bfr[c].v, (short)0, acc[r][c], false, false);

    __syncthreads();
  }

  // epilogue: C/D layout -> VGPR vr holds M = vr + 8*lane[4], N = lane[3:0]
#pragma unroll
  for (int r = 0; r < 2; ++r) {
    int mbase = m0 + wm * 32 + r * 16 + ((lane & 16) ? 8 : 0);
#pragma unroll
    for (int c = 0; c < 4; ++c) {
      int nn = n0 + wn * 64 + c * 16 + (lane & 15);
      if (nn < Nc) {
        float bv = bias[nn];
#pragma unroll
        for (int vr = 0; vr < 8; ++vr) {
          int gm = mbase + vr;
          if (gm < M) {
            float val = acc[r][c][vr] + bv;
            if (relu) val = fmaxf(val, 0.0f);
            C[(size_t)gm * Nc + nn] = val;
          }
        }
      }
    }
  }
}

// ---------------------------------------------------------------------------
// Weight convert + transpose: W f32 (K x Nc) -> Wt bf16 (Nc x K)
// ---------------------------------------------------------------------------
__global__ void convert_w_t(const float* __restrict__ W,
                            unsigned short* __restrict__ Wt, int K, int Nc)
{
  long long t = (long long)blockIdx.x * 256 + threadIdx.x;
  if (t >= (long long)K * Nc) return;
  int k = (int)(t % K);
  int n = (int)(t / K);
  Wt[(size_t)n * K + k] = f2bf(W[(size_t)k * Nc + n]);
}

// ---------------------------------------------------------------------------
// Gather / concat builders (f32 sources -> bf16 A matrices), 2 elems/thread.
// All feature dims are even, pairs never straddle a segment boundary.
// ---------------------------------------------------------------------------
__global__ void build_edge_A(unsigned short* __restrict__ A,
                             const float* __restrict__ x, int xdim,
                             const float* __restrict__ ea, int eadim,
                             const float* __restrict__ u, int udim,
                             const int* __restrict__ row,
                             const int* __restrict__ col,
                             const int* __restrict__ batch,
                             int E, int K)
{
  long long t = (long long)blockIdx.x * 256 + threadIdx.x;
  long long total = (long long)E * (K >> 1);
  if (t >= total) return;
  int e = (int)(t / (K >> 1));
  int k = (int)(t % (K >> 1)) * 2;
  const float* src; int off;
  if (k < xdim)                  { src = x  + (size_t)row[e] * xdim;            off = k; }
  else if (k < 2 * xdim)         { src = x  + (size_t)col[e] * xdim;            off = k - xdim; }
  else if (k < 2 * xdim + eadim) { src = ea + (size_t)e * eadim;                off = k - 2 * xdim; }
  else                           { src = u  + (size_t)batch[row[e]] * udim;     off = k - 2 * xdim - eadim; }
  A[(size_t)e * K + k]     = f2bf(src[off]);
  A[(size_t)e * K + k + 1] = f2bf(src[off + 1]);
}

__global__ void build_msg_A(unsigned short* __restrict__ A,
                            const float* __restrict__ x, int xdim,
                            const float* __restrict__ ea, int eadim,
                            const int* __restrict__ col,
                            int E, int K)
{
  long long t = (long long)blockIdx.x * 256 + threadIdx.x;
  long long total = (long long)E * (K >> 1);
  if (t >= total) return;
  int e = (int)(t / (K >> 1));
  int k = (int)(t % (K >> 1)) * 2;
  const float* src; int off;
  if (k < xdim) { src = x  + (size_t)col[e] * xdim; off = k; }
  else          { src = ea + (size_t)e * eadim;     off = k - xdim; }
  A[(size_t)e * K + k]     = f2bf(src[off]);
  A[(size_t)e * K + k + 1] = f2bf(src[off + 1]);
}

__global__ void build_node_A(unsigned short* __restrict__ A,
                             const float* __restrict__ x, int xdim,
                             const float* __restrict__ agg, int adim,
                             const float* __restrict__ u, int udim,
                             const int* __restrict__ batch,
                             int Nn, int K)
{
  long long t = (long long)blockIdx.x * 256 + threadIdx.x;
  long long total = (long long)Nn * (K >> 1);
  if (t >= total) return;
  int n = (int)(t / (K >> 1));
  int k = (int)(t % (K >> 1)) * 2;
  const float* src; int off;
  if (k < xdim)              { src = x   + (size_t)n * xdim;        off = k; }
  else if (k < xdim + adim)  { src = agg + (size_t)n * adim;        off = k - xdim; }
  else                       { src = u   + (size_t)batch[n] * udim; off = k - xdim - adim; }
  A[(size_t)n * K + k]     = f2bf(src[off]);
  A[(size_t)n * K + k + 1] = f2bf(src[off + 1]);
}

__global__ void build_glb_A(unsigned short* __restrict__ A,
                            const float* __restrict__ nagg, int ndim,
                            const float* __restrict__ eagg, int edim,
                            const float* __restrict__ u, int udim,
                            int G, int K)
{
  long long t = (long long)blockIdx.x * 256 + threadIdx.x;
  long long total = (long long)G * (K >> 1);
  if (t >= total) return;
  int g = (int)(t / (K >> 1));
  int k = (int)(t % (K >> 1)) * 2;
  const float* src; int off;
  if (k < ndim)              { src = nagg + (size_t)g * ndim; off = k; }
  else if (k < ndim + edim)  { src = eagg + (size_t)g * edim; off = k - ndim; }
  else                       { src = u    + (size_t)g * udim; off = k - ndim - edim; }
  A[(size_t)g * K + k]     = f2bf(src[off]);
  A[(size_t)g * K + k + 1] = f2bf(src[off + 1]);
}

// ---------------------------------------------------------------------------
// Aggregations
// ---------------------------------------------------------------------------
__global__ void zero_f32(float* __restrict__ p, long long n)
{
  long long i = (long long)blockIdx.x * 256 + threadIdx.x;
  if (i < n) p[i] = 0.0f;
}

__global__ void scatter_add(const float* __restrict__ msg,
                            const int* __restrict__ row,
                            float* __restrict__ agg, int E, int Cc)
{
  long long t = (long long)blockIdx.x * 256 + threadIdx.x;
  if (t >= (long long)E * Cc) return;
  int e = (int)(t / Cc);
  int c = (int)(t % Cc);
  atomicAdd(&agg[(size_t)row[e] * Cc + c], msg[(size_t)e * Cc + c]);
}

__global__ void eagg_atomic(const float* __restrict__ ea,
                            const int* __restrict__ row,
                            const int* __restrict__ batch,
                            float* __restrict__ eagg, int E, int Cc)
{
  long long t = (long long)blockIdx.x * 256 + threadIdx.x;
  if (t >= (long long)E * Cc) return;
  int e = (int)(t / Cc);
  int c = (int)(t % Cc);
  atomicAdd(&eagg[(size_t)batch[row[e]] * Cc + c], ea[(size_t)e * Cc + c]);
}

// batch is sorted, nodes_per_graph contiguous nodes per graph
__global__ void nagg_sum(const float* __restrict__ x, float* __restrict__ out,
                         int Cc, int npg)
{
  int g = blockIdx.x;
  int c = blockIdx.y * 256 + threadIdx.x;
  if (c >= Cc) return;
  float s = 0.0f;
  for (int i = 0; i < npg; ++i)
    s += x[(size_t)(g * npg + i) * Cc + c];
  out[(size_t)g * Cc + c] = s;
}

// ---------------------------------------------------------------------------
// BatchNorm (training mode, biased variance)
// ---------------------------------------------------------------------------
__global__ void bn_stats(const float* __restrict__ h, int M, int Cc,
                         float* __restrict__ mean, float* __restrict__ rstd)
{
  __shared__ float sh[256], sh2[256];
  int c = blockIdx.x;
  float s = 0.0f, s2 = 0.0f;
  for (int i = threadIdx.x; i < M; i += 256) {
    float v = h[(size_t)i * Cc + c];
    s += v; s2 += v * v;
  }
  sh[threadIdx.x] = s; sh2[threadIdx.x] = s2;
  __syncthreads();
  for (int o = 128; o > 0; o >>= 1) {
    if (threadIdx.x < o) { sh[threadIdx.x] += sh[threadIdx.x + o]; sh2[threadIdx.x] += sh2[threadIdx.x + o]; }
    __syncthreads();
  }
  if (threadIdx.x == 0) {
    float m = sh[0] / (float)M;
    float v = sh2[0] / (float)M - m * m;
    mean[c] = m;
    rstd[c] = rsqrtf(v + 1e-5f);
  }
}

__global__ void bn_apply(float* __restrict__ h, long long M, int Cc,
                         const float* __restrict__ mean,
                         const float* __restrict__ rstd,
                         const float* __restrict__ g,
                         const float* __restrict__ b)
{
  long long t = (long long)blockIdx.x * 256 + threadIdx.x;
  if (t >= M * Cc) return;
  int c = (int)(t % Cc);
  h[t] = (h[t] - mean[c]) * rstd[c] * g[c] + b[c];
}

// ---------------------------------------------------------------------------
// Host orchestration
// ---------------------------------------------------------------------------
extern "C" void kernel_launch(void* const* d_in, const int* in_sizes, int n_in,
                              void* d_out, int out_size, void* d_ws, size_t ws_size,
                              hipStream_t stream)
{
  (void)in_sizes; (void)n_in; (void)out_size; (void)ws_size;

  constexpr int N = 4096, E = 65536, G = 64;
  constexpr int NPG = N / G;   // 64 nodes per graph, sorted batch

  const float* in_x  = (const float*)d_in[0];
  const int*   eidx  = (const int*)d_in[1];
  const int*   row   = eidx;       // dest
  const int*   col   = eidx + E;   // src
  const float* in_ea = (const float*)d_in[2];
  const float* in_u  = (const float*)d_in[3];
  const int*   batch = (const int*)d_in[4];

  // Param unpack — jax pytree (list of dicts, keys sorted; capitals < lowercase)
  struct LP {
    const float *eW, *eb, *gW, *gb, *n1W, *n1b, *n2W, *n2b;
    const float *bne_b, *bne_g, *bnu_b, *bnu_g, *bnx_b, *bnx_g;
  } L[4];
  {
    int idx = 5;
    for (int l = 0; l < 4; ++l) {
      if (l < 3) {
        L[l].bne_b = (const float*)d_in[idx++]; L[l].bne_g = (const float*)d_in[idx++];
        L[l].bnu_b = (const float*)d_in[idx++]; L[l].bnu_g = (const float*)d_in[idx++];
        L[l].bnx_b = (const float*)d_in[idx++]; L[l].bnx_g = (const float*)d_in[idx++];
      }
      L[l].eW  = (const float*)d_in[idx++]; L[l].eb  = (const float*)d_in[idx++];
      L[l].gW  = (const float*)d_in[idx++]; L[l].gb  = (const float*)d_in[idx++];
      L[l].n1W = (const float*)d_in[idx++]; L[l].n1b = (const float*)d_in[idx++];
      L[l].n2W = (const float*)d_in[idx++]; L[l].n2b = (const float*)d_in[idx++];
    }
  }

  // Workspace carve-up (256B aligned)
  char* wp = (char*)d_ws;
  auto alloc = [&](size_t bytes) -> void* {
    void* p = (void*)wp;
    wp += (bytes + 255) & ~(size_t)255;
    return p;
  };
  float* ea_buf = (float*)alloc((size_t)E * 256 * 4);
  float* x_buf  = (float*)alloc((size_t)N * 256 * 4);
  float* u_buf  = (float*)alloc((size_t)G * 256 * 4);
  unsigned short* Ae = (unsigned short*)alloc((size_t)E * 1024 * 2); // reused as A_m
  unsigned short* An = (unsigned short*)alloc((size_t)N * 1024 * 2);
  unsigned short* Ag = (unsigned short*)alloc((size_t)G * 1024 * 2);
  float* msg  = (float*)alloc((size_t)E * 256 * 4);
  float* agg  = (float*)alloc((size_t)N * 256 * 4);
  float* nagg = (float*)alloc((size_t)G * 256 * 4);
  float* eagg = (float*)alloc((size_t)G * 256 * 4);
  unsigned short* Wt = (unsigned short*)alloc((size_t)1024 * 256 * 2);
  float* bn_m = (float*)alloc(256 * 4);
  float* bn_r = (float*)alloc(256 * 4);

  float* out_x  = (float*)d_out;                       // N x 128
  float* out_ea = out_x + (size_t)N * 128;             // E x 128
  float* out_u  = out_ea + (size_t)E * 128;            // G x 128

  auto gemm = [&](const unsigned short* A, const float* W, const float* bias,
                  float* Cout, int M, int Nc, int K, int relu) {
    long long wtot = (long long)K * Nc;
    convert_w_t<<<dim3((unsigned)CDIV(wtot, 256)), dim3(256), 0, stream>>>(W, Wt, K, Nc);
    dim3 grid((unsigned)CDIV(Nc, 128), (unsigned)CDIV(M, 128));
    gemm_bf16_wmma<<<grid, dim3(256), 0, stream>>>(A, Wt, bias, Cout, M, Nc, K, relu);
  };

  for (int l = 0; l < 4; ++l) {
    const int act = (l < 3) ? 1 : 0;
    const float* xs  = (l == 0) ? in_x  : x_buf;  const int xdim  = (l == 0) ? 128 : 256;
    const float* eas = (l == 0) ? in_ea : ea_buf; const int eadim = (l == 0) ? 64  : 256;
    const float* us  = (l == 0) ? in_u  : u_buf;  const int udim  = (l == 0) ? 64  : 256;
    const int e_out = (l < 3) ? 256 : 128;
    const int n_out = (l < 3) ? 256 : 128;
    const int g_out = (l < 3) ? 256 : 128;
    float* ea_dst = (l == 3) ? out_ea : ea_buf;
    float* x_dst  = (l == 3) ? out_x  : x_buf;
    float* u_dst  = (l == 3) ? out_u  : u_buf;

    // ---- EdgeModel: concat[x[row], x[col], ea, u[batch[row]]] -> Linear(+ReLU)
    const int Ke = 2 * xdim + eadim + udim;
    {
      long long tot = (long long)E * (Ke >> 1);
      build_edge_A<<<dim3((unsigned)CDIV(tot, 256)), dim3(256), 0, stream>>>(
          Ae, xs, xdim, eas, eadim, us, udim, row, col, batch, E, Ke);
    }
    gemm(Ae, L[l].eW, L[l].eb, ea_dst, E, e_out, Ke, act);

    // ---- NodeModel part 1: msg = MLP(concat[x[col], ea]); agg = scatter_sum(msg, row)
    const int Km = xdim + e_out;
    {
      long long tot = (long long)E * (Km >> 1);
      build_msg_A<<<dim3((unsigned)CDIV(tot, 256)), dim3(256), 0, stream>>>(
          Ae, xs, xdim, ea_dst, e_out, col, E, Km);
    }
    gemm(Ae, L[l].n1W, L[l].n1b, msg, E, n_out, Km, act);
    {
      long long na = (long long)N * n_out;
      zero_f32<<<dim3((unsigned)CDIV(na, 256)), dim3(256), 0, stream>>>(agg, na);
      long long tot = (long long)E * n_out;
      scatter_add<<<dim3((unsigned)CDIV(tot, 256)), dim3(256), 0, stream>>>(msg, row, agg, E, n_out);
    }

    // ---- NodeModel part 2: x = MLP(concat[x, agg, u[batch]])
    const int Kn = xdim + n_out + udim;
    {
      long long tot = (long long)N * (Kn >> 1);
      build_node_A<<<dim3((unsigned)CDIV(tot, 256)), dim3(256), 0, stream>>>(
          An, xs, xdim, agg, n_out, us, udim, batch, N, Kn);
    }
    gemm(An, L[l].n2W, L[l].n2b, x_dst, N, n_out, Kn, act);

    // ---- GlobalModel: u = MLP(concat[nagg, eagg, u])
    nagg_sum<<<dim3(G, (unsigned)CDIV(n_out, 256)), dim3(256), 0, stream>>>(x_dst, nagg, n_out, NPG);
    {
      long long ga = (long long)G * e_out;
      zero_f32<<<dim3((unsigned)CDIV(ga, 256)), dim3(256), 0, stream>>>(eagg, ga);
      long long tot = (long long)E * e_out;
      eagg_atomic<<<dim3((unsigned)CDIV(tot, 256)), dim3(256), 0, stream>>>(ea_dst, row, batch, eagg, E, e_out);
    }
    const int Kg = n_out + e_out + udim;
    {
      long long tot = (long long)G * (Kg >> 1);
      build_glb_A<<<dim3((unsigned)CDIV(tot, 256)), dim3(256), 0, stream>>>(
          Ag, nagg, n_out, eagg, e_out, us, udim, G, Kg);
    }
    gemm(Ag, L[l].gW, L[l].gb, u_dst, G, g_out, Kg, act);

    // ---- BatchNorm on x / ea / u (layers 0..2)
    if (l < 3) {
      bn_stats<<<dim3(256), dim3(256), 0, stream>>>(x_buf, N, 256, bn_m, bn_r);
      bn_apply<<<dim3((unsigned)CDIV((long long)N * 256, 256)), dim3(256), 0, stream>>>(
          x_buf, N, 256, bn_m, bn_r, L[l].bnx_g, L[l].bnx_b);
      bn_stats<<<dim3(256), dim3(256), 0, stream>>>(ea_buf, E, 256, bn_m, bn_r);
      bn_apply<<<dim3((unsigned)CDIV((long long)E * 256, 256)), dim3(256), 0, stream>>>(
          ea_buf, E, 256, bn_m, bn_r, L[l].bne_g, L[l].bne_b);
      bn_stats<<<dim3(256), dim3(256), 0, stream>>>(u_buf, G, 256, bn_m, bn_r);
      bn_apply<<<dim3((unsigned)CDIV((long long)G * 256, 256)), dim3(256), 0, stream>>>(
          u_buf, G, 256, bn_m, bn_r, L[l].bnu_g, L[l].bnu_b);
    }
  }
}
